// LSTMBySplit_31610959299232
// MI455X (gfx1250) — matrix operand order repaired and checked by reading, not compiled, not run on hardware
//
#include <hip/hip_runtime.h>
#include <cstdint>

typedef __attribute__((ext_vector_type(8)))  __bf16 v8bf;
typedef __attribute__((ext_vector_type(16))) __bf16 v16bf;
typedef __attribute__((ext_vector_type(8)))  float  v8f;

#define T_STEPS 64
#define BATCH   1024
#define HDIM    512
#define KTOT    512
#define NLAYERS 10
#define NCHUNK  32            // 512 k (X) + 512 k (H) in 32-k chunks

__device__ __forceinline__ float sigmoidf_(float x) {
    return 1.0f / (1.0f + __expf(-x));
}

// Async global -> LDS copy of 16 bytes (CDNA5, tracked by ASYNCcnt).
__device__ __forceinline__ void async_ld_b128(uint32_t lds_off,
                                              const void* base,
                                              uint32_t byte_off) {
    asm volatile("global_load_async_to_lds_b128 %0, %1, %2"
                 :: "v"(lds_off), "v"(byte_off), "s"(base) : "memory");
}
__device__ __forceinline__ void async_ld_b128_o16(uint32_t lds_off,
                                                  const void* base,
                                                  uint32_t byte_off) {
    asm volatile("global_load_async_to_lds_b128 %0, %1, %2 offset:16"
                 :: "v"(lds_off), "v"(byte_off), "s"(base) : "memory");
}
__device__ __forceinline__ void wait_asynccnt0() {
    asm volatile("s_wait_asynccnt 0x0" ::: "memory");
}
// Leave the 4 most recently issued async ops (newest chunk) in flight.
__device__ __forceinline__ void wait_asynccnt4() {
    asm volatile("s_wait_asynccnt 0x4" ::: "memory");
}

// ---------------------------------------------------------------------------
// One LSTM cell: gates[B,4H] = X[B,K]*Wih + Hprev[B,K]*Whh (+bias), fused
// pointwise update.
// Block tile: M=128 rows of B, N = 4 gates x 32 h-cols. 8 waves.
// Wave w -> rows (w>>1)*32 + {0,16}, hi = w&1 (16 h-cols).
// 2 M-tiles x 4 gates = 8 WMMA per k-chunk per wave; all four gate values of
// each (b,h) element share a lane, so the LSTM update is fully in-register.
// 3-deep async pipeline with compile-time buffer rotation: slot s of the
// kq-loop issues chunk kq+2+s into a literal buffer index, computes chunk
// kq+s, then s_wait_asynccnt 4 guarantees the next chunk has landed while the
// newest stays in flight. Last two chunks are peeled (no new issues).
// ---------------------------------------------------------------------------
__global__ __launch_bounds__(256) void lstm_cell_kernel(
    const __bf16* __restrict__ x_src,        // [B][K] bf16 (layer input)
    const __bf16* __restrict__ h_prev,       // [B][K] bf16
    const __bf16* __restrict__ Wih,          // [4][H][K] bf16 (transposed)
    const __bf16* __restrict__ Whh,          // [4][H][K] bf16
    const float*  __restrict__ bsum,         // [4][H] fp32 (b_ih + b_hh)
    float*        __restrict__ c_io,         // [B][H] fp32, in-place
    __bf16*       __restrict__ h_out,        // [B][H] bf16
    float*        __restrict__ final_out)    // [B][H] fp32 or nullptr
{
    __shared__ __align__(16) __bf16 ldsA[3][128][40];   // 128 rows x 32k (+pad)
    __shared__ __align__(16) __bf16 ldsB[3][128][40];   // 128 cols x 32k (+pad)

    const int tid  = threadIdx.x;
    const int lane = tid & 31;
    const int wave = tid >> 5;
    const int lg   = lane >> 4;        // lane group 0/1
    const int lm   = lane & 15;
    const int rowb = (wave >> 1) * 32; // wave's 32-row base within block tile
    const int hi   = wave & 1;         // 16-col h sub-tile
    const int b0   = blockIdx.x * 128;
    const int h0   = blockIdx.y * 32;

    // Loaders: 256 threads move two 128x32 bf16 tiles (32 B per thread each)
    const int l_row = tid >> 1;          // 0..127 (A row / B col)
    const int l_ko  = (tid & 1) * 16;    // k-half within 32-k chunk
    const int l_g   = l_row >> 5;        // B: gate
    const int l_h   = l_row & 31;        // B: h within block

    const v8f vzero = {0.f, 0.f, 0.f, 0.f, 0.f, 0.f, 0.f, 0.f};
    v8f acc[2][4] = {{vzero, vzero, vzero, vzero},
                     {vzero, vzero, vzero, vzero}};

    // Per-thread invariant pieces of the addresses
    const uint32_t a_row_byte = (uint32_t)((b0 + l_row) * KTOT + l_ko) * 2u;
    const uint32_t w_row_byte = (uint32_t)(((l_g * HDIM) + (h0 + l_h)) * KTOT + l_ko) * 2u;

    auto issue_chunk = [&](int kc, int nb) {
        const int phase = kc >> 4;           // 0: X*Wih, 1: H*Whh
        const uint32_t kbb = (uint32_t)((kc & 15) * 32) * 2u;  // k byte offset
        const void* abase = phase ? (const void*)h_prev : (const void*)x_src;
        const void* wbase = phase ? (const void*)Whh : (const void*)Wih;
        const uint32_t aldso = (uint32_t)(uintptr_t)&ldsA[nb][l_row][l_ko];
        const uint32_t wldso = (uint32_t)(uintptr_t)&ldsB[nb][l_row][l_ko];
        async_ld_b128    (aldso, abase, a_row_byte + kbb);
        async_ld_b128_o16(aldso, abase, a_row_byte + kbb);
        async_ld_b128    (wldso, wbase, w_row_byte + kbb);
        async_ld_b128_o16(wldso, wbase, w_row_byte + kbb);
    };

    // 8 WMMA on the chunk resident in buffer `cur` (literal constant).
    auto compute_chunk = [&](int cur) {
        // A fragments (two M tiles). Documented 16-bit A layout:
        //   e=0..7  -> k = 8*lg + e        (bytes 16*lg)
        //   e=8..15 -> k = 16 + 8*lg + e-8 (bytes 32 + 16*lg)
        v16bf af[2];
#pragma unroll
        for (int m2 = 0; m2 < 2; ++m2) {
            const __bf16* pa = &ldsA[cur][rowb + m2 * 16 + lm][8 * lg];
            v8bf alo = *(const v8bf*)pa;
            v8bf ahi = *(const v8bf*)(pa + 16);
            af[m2] = __builtin_shufflevector(alo, ahi,
                0, 1, 2, 3, 4, 5, 6, 7, 8, 9, 10, 11, 12, 13, 14, 15);
        }
#pragma unroll
        for (int g = 0; g < 4; ++g) {
            // B fragment: col n = lm; same k layout as A (8*lg + e, 16 + 8*lg + e-8)
            const __bf16* pb = &ldsB[cur][g * 32 + hi * 16 + lm][8 * lg];
            v8bf blo = *(const v8bf*)pb;
            v8bf bhi = *(const v8bf*)(pb + 16);
            v16bf bfr = __builtin_shufflevector(blo, bhi,
                0, 1, 2, 3, 4, 5, 6, 7, 8, 9, 10, 11, 12, 13, 14, 15);
#pragma unroll
            for (int m2 = 0; m2 < 2; ++m2) {
                acc[m2][g] = __builtin_amdgcn_wmma_f32_16x16x32_bf16(
                    false, af[m2], false, bfr, (short)0, acc[m2][g], false, false);
            }
        }
    };

    issue_chunk(0, 0);
    issue_chunk(1, 1);
    wait_asynccnt4();         // chunk 0 resident, chunk 1 still in flight
    __syncthreads();

    // Steady state: 30 chunks, compile-time 3-slot buffer rotation.
#pragma unroll 1
    for (int kq = 0; kq < NCHUNK - 2; kq += 3) {
        issue_chunk(kq + 2, 2); compute_chunk(0);
        wait_asynccnt4(); __syncthreads();
        issue_chunk(kq + 3, 0); compute_chunk(1);
        wait_asynccnt4(); __syncthreads();
        issue_chunk(kq + 4, 1); compute_chunk(2);
        wait_asynccnt4(); __syncthreads();
    }
    // Peeled tail: chunks 30 (buf 0) and 31 (buf 1), nothing left to issue.
    compute_chunk(0);
    wait_asynccnt0(); __syncthreads();
    compute_chunk(1);

    // ---- fused LSTM pointwise update ----
    // D layout: lane -> n = lm; VGPR r -> m = r + 8*lg
    const int h_gl = h0 + hi * 16 + lm;
    const float bi = bsum[0 * HDIM + h_gl];
    const float bf = bsum[1 * HDIM + h_gl];
    const float bc = bsum[2 * HDIM + h_gl];
    const float bo = bsum[3 * HDIM + h_gl];

#pragma unroll
    for (int m2 = 0; m2 < 2; ++m2) {
#pragma unroll
        for (int r = 0; r < 8; ++r) {
            const int brow = b0 + rowb + m2 * 16 + lg * 8 + r;
            const size_t idx = (size_t)brow * HDIM + (size_t)h_gl;
            const float gi = sigmoidf_(acc[m2][0][r] + bi);
            const float gf = sigmoidf_(acc[m2][1][r] + bf);
            const float gc = tanhf(acc[m2][2][r] + bc);
            const float go = sigmoidf_(acc[m2][3][r] + bo);
            const float cn = gf * c_io[idx] + gi * gc;
            c_io[idx] = cn;
            const float hv = go * tanhf(cn);
            h_out[idx] = (__bf16)hv;
            if (final_out) final_out[idx] = hv;
        }
    }
}

// ---------------------------------------------------------------------------
// Prep kernels
// ---------------------------------------------------------------------------
// w: [L*4][K][H] fp32  ->  wt: [L*4][H][K] bf16
__global__ void prep_weights_kernel(const float* __restrict__ w,
                                    __bf16* __restrict__ wt, long n) {
    long i = (long)blockIdx.x * blockDim.x + threadIdx.x;
    if (i >= n) return;
    long m   = i / ((long)KTOT * HDIM);
    long rem = i - m * ((long)KTOT * HDIM);
    long h   = rem / KTOT;
    long k   = rem - h * KTOT;
    wt[i] = (__bf16)w[(m * KTOT + k) * HDIM + h];
}

// x fp32 -> bf16 straight copy
__global__ void prep_x_kernel(const float* __restrict__ x,
                              __bf16* __restrict__ xb, long n) {
    long i = (long)blockIdx.x * blockDim.x + threadIdx.x;
    if (i < n) xb[i] = (__bf16)x[i];
}

__global__ void prep_bias_kernel(const float* __restrict__ bih,
                                 const float* __restrict__ bhh,
                                 float* __restrict__ bs, int n) {
    int i = blockIdx.x * blockDim.x + threadIdx.x;
    if (i < n) bs[i] = bih[i] + bhh[i];
}

__global__ void zero_state_kernel(__bf16* __restrict__ h,
                                  float* __restrict__ c, long n) {
    long i = (long)blockIdx.x * blockDim.x + threadIdx.x;
    if (i < n) { h[i] = (__bf16)0.0f; c[i] = 0.0f; }
}

// ---------------------------------------------------------------------------
extern "C" void kernel_launch(void* const* d_in, const int* in_sizes, int n_in,
                              void* d_out, int out_size, void* d_ws, size_t ws_size,
                              hipStream_t stream) {
    const float* x   = (const float*)d_in[0];   // [T][B][IN] fp32
    const float* wih = (const float*)d_in[1];   // [L][4][IN][H]
    const float* whh = (const float*)d_in[2];   // [L][4][H][H]
    const float* bih = (const float*)d_in[3];   // [L][4][H]
    const float* bhh = (const float*)d_in[4];   // [L][4][H]
    float* out = (float*)d_out;                  // [B][H]

    char* p = (char*)d_ws;
    auto take = [&](size_t bytes) -> char* {
        char* r = p;
        p += (bytes + 255) & ~(size_t)255;
        return r;
    };
    const size_t WSZ = (size_t)NLAYERS * 4 * HDIM * KTOT;   // weight elements
    const size_t SSZ = (size_t)NLAYERS * BATCH * HDIM;      // state elements
    const size_t XSZ = (size_t)T_STEPS * BATCH * KTOT;      // input elements
    __bf16* WihT = (__bf16*)take(WSZ * 2);
    __bf16* WhhT = (__bf16*)take(WSZ * 2);
    __bf16* Xbf  = (__bf16*)take(XSZ * 2);
    float*  Bsum = (float*)take((size_t)NLAYERS * 4 * HDIM * 4);
    __bf16* Ha   = (__bf16*)take(SSZ * 2);
    __bf16* Hb   = (__bf16*)take(SSZ * 2);
    float*  C    = (float*)take(SSZ * 4);

    {
        long n = (long)WSZ;
        unsigned grid = (unsigned)((n + 255) / 256);
        prep_weights_kernel<<<grid, 256, 0, stream>>>(wih, WihT, n);
        prep_weights_kernel<<<grid, 256, 0, stream>>>(whh, WhhT, n);
    }
    {
        long n = (long)XSZ;
        prep_x_kernel<<<(unsigned)((n + 255) / 256), 256, 0, stream>>>(x, Xbf, n);
    }
    {
        int n = NLAYERS * 4 * HDIM;
        prep_bias_kernel<<<(n + 255) / 256, 256, 0, stream>>>(bih, bhh, Bsum, n);
    }
    {
        long n = (long)SSZ;
        zero_state_kernel<<<(unsigned)((n + 255) / 256), 256, 0, stream>>>(Ha, C, n);
    }

    dim3 grid(BATCH / 128, HDIM / 32);   // 8 x 16 blocks
    __bf16* hin  = Ha;
    __bf16* hout = Hb;
    for (int t = 0; t < T_STEPS; ++t) {
        for (int j = 0; j < NLAYERS; ++j) {
            const __bf16* xs = (j == 0)
                ? (Xbf + (size_t)t * BATCH * KTOT)
                : (hout + (size_t)(j - 1) * BATCH * HDIM);
            float* fo = (t == T_STEPS - 1 && j == NLAYERS - 1) ? out : nullptr;
            lstm_cell_kernel<<<grid, 256, 0, stream>>>(
                xs,
                hin  + (size_t)j * BATCH * HDIM,
                WihT + (size_t)j * 4 * HDIM * KTOT,
                WhhT + (size_t)j * 4 * HDIM * KTOT,
                Bsum + (size_t)j * 4 * HDIM,
                C    + (size_t)j * BATCH * HDIM,
                hout + (size_t)j * BATCH * HDIM,
                fo);
        }
        __bf16* tmp = hin; hin = hout; hout = tmp;
    }
}
